// PicoDetPostProcessor_79731772883191
// MI455X (gfx1250) — matrix operand order repaired and checked by reading, not compiled, not run on hardware
//
#include <hip/hip_runtime.h>
#include <math.h>

typedef __attribute__((ext_vector_type(16))) _Float16 v16h;
typedef __attribute__((ext_vector_type(8)))  float    v8f;

#define NUM_CLASSES 1203
#define NMS_PRE     1000
#define K_TOTAL     4000
#define MAX_DET     100
#define OVERCAP     4096
#define SCORE_THR   0.025f
#define IOU_THR     0.6f
// logit(0.025) = log(0.025/0.975)
#define LOGIT_THR  (-3.6635616461296463f)

__device__ __forceinline__ unsigned f2key(float f) {
    unsigned u = __float_as_uint(f);
    return (u & 0x80000000u) ? ~u : (u | 0x80000000u);
}
__device__ __forceinline__ float key2f(unsigned key) {
    unsigned u = (key & 0x80000000u) ? (key ^ 0x80000000u) : ~key;
    return __uint_as_float(u);
}

// ---------------- init: zero histograms + counters ----------------
__global__ void init_kernel(unsigned* hist, int* meta) {
    int i = blockIdx.x * 256 + threadIdx.x;
    if (i < 4 * 2048) hist[i] = 0u;
    if (i < 32) meta[i] = 0;
}

// ---------------- pass 1: per-level 2048-bin histogram of logit keys ----------------
__global__ __launch_bounds__(256)
void hist_kernel(const float* __restrict__ x, int n4, unsigned* __restrict__ ghist) {
    __shared__ unsigned lh[2048];
    for (int i = threadIdx.x; i < 2048; i += 256) lh[i] = 0u;
    __syncthreads();
    const int stride = gridDim.x * 256;
    const float4* x4 = (const float4*)x;
    for (int i = blockIdx.x * 256 + threadIdx.x; i < n4; i += stride) {
        __builtin_prefetch(x4 + i + stride, 0, 0);   // global_prefetch_b8
        float4 v = x4[i];
        float vv[4] = {v.x, v.y, v.z, v.w};
#pragma unroll
        for (int c = 0; c < 4; ++c) {
            float f = vv[c];
            if (f > LOGIT_THR) atomicAdd(&lh[f2key(f) >> 21], 1u);
        }
    }
    __syncthreads();
    for (int i = threadIdx.x; i < 2048; i += 256) {
        unsigned cnt = lh[i];
        if (cnt) atomicAdd(&ghist[i], cnt);
    }
}

// ---------------- pass 2: descending scan -> cutoff bin per level ----------------
// meta layout: [0..3]=cntA(atomic) [4..7]=cntB(atomic) [8..11]=cutBin [12..15]=cntAbove [16..19]=filled
__global__ void scan_kernel(const unsigned* __restrict__ hist, int* __restrict__ meta) {
    int l = threadIdx.x;
    if (l < 4) {
        const unsigned* h = hist + l * 2048;
        unsigned cum = 0; int cut = -1; unsigned above = 0;
        for (int b = 2047; b >= 0; --b) {
            unsigned nb = cum + h[b];
            if (nb >= (unsigned)NMS_PRE) { cut = b; above = cum; break; }
            cum = nb;
        }
        if (cut < 0) above = cum;  // total < 1000: take everything
        meta[8 + l]  = cut;
        meta[12 + l] = (int)above;
    }
}

// ---------------- pass 3: compaction ----------------
__global__ __launch_bounds__(256)
void compact_kernel(const float* __restrict__ x, int n4, int HW, int level,
                    int* __restrict__ meta, int* __restrict__ selIdx,
                    float* __restrict__ selVal, unsigned long long* __restrict__ ovKey) {
    const int cut = meta[8 + level];
    const int stride = gridDim.x * 256;
    const float4* x4 = (const float4*)x;
    for (int i = blockIdx.x * 256 + threadIdx.x; i < n4; i += stride) {
        float4 v = x4[i];
        float vv[4] = {v.x, v.y, v.z, v.w};
#pragma unroll
        for (int c = 0; c < 4; ++c) {
            float f = vv[c];
            if (f > LOGIT_THR) {
                unsigned key = f2key(f);
                int bin = (int)(key >> 21);
                if (bin > cut) {
                    int pos = atomicAdd(&meta[level], 1);
                    if (pos < NMS_PRE) {
                        int midx = 4 * i + c;                 // (C,H,W) memory order
                        int ch = midx / HW, pt = midx - ch * HW;
                        int refIdx = pt * NUM_CLASSES + ch;   // (H,W,C) reference order
                        selIdx[level * NMS_PRE + pos] = refIdx;
                        selVal[level * NMS_PRE + pos] = f;
                    }
                } else if (bin == cut) {
                    int pos = atomicAdd(&meta[4 + level], 1);
                    if (pos < OVERCAP) {
                        int midx = 4 * i + c;
                        int ch = midx / HW, pt = midx - ch * HW;
                        int refIdx = pt * NUM_CLASSES + ch;
                        ovKey[level * OVERCAP + pos] =
                            ((unsigned long long)key << 32) | (unsigned)(~(unsigned)refIdx);
                    }
                }
            }
        }
    }
}

// ---------------- pass 4: exact tie resolution in the cutoff bin ----------------
__global__ __launch_bounds__(1024)
void fixup_kernel(int* __restrict__ meta, const unsigned long long* __restrict__ ovKeyG,
                  int* __restrict__ selIdx, float* __restrict__ selVal) {
    __shared__ unsigned long long s[OVERCAP];
    const int l = blockIdx.x, tid = threadIdx.x;
    int M = meta[4 + l]; if (M > OVERCAP) M = OVERCAP;
    const unsigned long long* ov = ovKeyG + l * OVERCAP;
    for (int i = tid; i < OVERCAP; i += 1024) s[i] = (i < M) ? ov[i] : 0ull;
    __syncthreads();
    // bitonic sort, descending (ties resolved toward smaller refIdx via ~idx in low bits)
    for (unsigned k = 2; k <= OVERCAP; k <<= 1) {
        for (unsigned j = k >> 1; j > 0; j >>= 1) {
            for (unsigned i = tid; i < OVERCAP; i += 1024) {
                unsigned ixj = i ^ j;
                if (ixj > i) {
                    unsigned long long A = s[i], B = s[ixj];
                    bool up = ((i & k) == 0);
                    if (up ? (A < B) : (A > B)) { s[i] = B; s[ixj] = A; }
                }
            }
            __syncthreads();
        }
    }
    int cA = meta[l]; if (cA > NMS_PRE) cA = NMS_PRE;
    int cut = meta[8 + l];
    int need = (cut >= 0) ? (NMS_PRE - cA) : 0;
    if (need < 0) need = 0;
    int take = need < M ? need : M;
    for (int i = tid; i < take; i += 1024) {
        unsigned long long v = s[i];
        unsigned key = (unsigned)(v >> 32);
        int refIdx = (int)(~(unsigned)(v & 0xffffffffu));
        selIdx[l * NMS_PRE + cA + i] = refIdx;
        selVal[l * NMS_PRE + cA + i] = key2f(key);
    }
    if (tid == 0) meta[16 + l] = cA + take;
}

// ---------------- pass 5: DFL decode via V_WMMA_F32_16X16X32_F16 ----------------
// Each wave handles 16 candidates. A(16x32 f16) = 4 softmax-8 distributions per row;
// B(32x16 f16) = block-diagonal arange(8) weights -> D[m][n] = expected bin (n=l,t,r,b).
__global__ __launch_bounds__(256)
void decode_kernel(const int* __restrict__ selIdx, const float* __restrict__ selVal,
                   const int* __restrict__ meta,
                   const float* __restrict__ bb0, const float* __restrict__ bb1,
                   const float* __restrict__ bb2, const float* __restrict__ bb3,
                   float* __restrict__ boxes, float* __restrict__ scores,
                   int* __restrict__ labels) {
    __shared__ float dls[8 * 256];   // 8 waves x (16x16 f32 D tile)
    const int tid = threadIdx.x, wave = tid >> 5, lane = tid & 31;
    const int slot = blockIdx.x * 128 + wave * 16 + (lane & 15);
    int level = slot / NMS_PRE; if (level > 3) level = 3;
    const int li = slot - level * NMS_PRE;
    const bool inRange = slot < K_TOTAL;

    int Wl, strideL; const float* bb;
    switch (level) {
        case 0:  Wl = 160; strideL = 8;  bb = bb0; break;
        case 1:  Wl = 80;  strideL = 16; bb = bb1; break;
        case 2:  Wl = 40;  strideL = 32; bb = bb2; break;
        default: Wl = 20;  strideL = 64; bb = bb3; break;
    }
    const int HWl = Wl * Wl;
    const int filled = meta[16 + level];
    const bool valid = inRange && (li < filled);
    const int idx = valid ? selIdx[level * NMS_PRE + li] : 0;
    const float logit = valid ? selVal[level * NMS_PRE + li] : 0.f;
    const int pt = idx / NUM_CLASSES;
    const int lab = idx - pt * NUM_CLASSES;

    // lanes 0-15 supply dists {0,1}; lanes 16-31 supply dists {2,3} for the same row
    const int part = lane >> 4;
    float p[2][8];
#pragma unroll
    for (int h = 0; h < 2; ++h) {
        const int dch = 2 * part + h;
        float v[8];
#pragma unroll
        for (int t = 0; t < 8; ++t)
            v[t] = valid ? bb[(dch * 8 + t) * HWl + pt] : 0.f;
        float mx = v[0];
#pragma unroll
        for (int t = 1; t < 8; ++t) mx = fmaxf(mx, v[t]);
        float sum = 0.f;
#pragma unroll
        for (int t = 0; t < 8; ++t) { v[t] = expf(v[t] - mx); sum += v[t]; }
        float inv = 1.f / sum;
#pragma unroll
        for (int t = 0; t < 8; ++t) p[h][t] = v[t] * inv;
    }
    v16h a;
#pragma unroll
    for (int t = 0; t < 8; ++t) { a[t] = (_Float16)p[0][t]; a[8 + t] = (_Float16)p[1][t]; }
    // B: lane = column N (mod 16); halves = K group (0-15 for lanes<16, 16-31 otherwise)
    v16h bm;
    const int col = lane & 15;
    const int kbase = (lane < 16) ? 0 : 16;
#pragma unroll
    for (int h = 0; h < 16; ++h) {
        int k = kbase + h;
        bm[h] = ((k >> 3) == col) ? (_Float16)(float)(k & 7) : (_Float16)0.f;
    }
    v8f c = {};
    c = __builtin_amdgcn_wmma_f32_16x16x32_f16(false, a, false, bm, (short)0, c, false, false);

    const int row0 = (lane < 16) ? 0 : 8;
#pragma unroll
    for (int r = 0; r < 8; ++r) dls[wave * 256 + (row0 + r) * 16 + col] = c[r];
    __syncthreads();

    if (lane < 16 && inRange) {
        const float s = (float)strideL;
        const float dl = dls[wave * 256 + lane * 16 + 0] * s;
        const float dt = dls[wave * 256 + lane * 16 + 1] * s;
        const float dr = dls[wave * 256 + lane * 16 + 2] * s;
        const float db = dls[wave * 256 + lane * 16 + 3] * s;
        const float px = ((float)(pt % Wl) + 0.5f) * s;
        const float py = ((float)(pt / Wl) + 0.5f) * s;
        if (valid) {
            boxes[slot * 4 + 0] = px - dl;
            boxes[slot * 4 + 1] = py - dt;
            boxes[slot * 4 + 2] = px + dr;
            boxes[slot * 4 + 3] = py + db;
            scores[slot] = 1.f / (1.f + expf(-logit));
            labels[slot] = lab;
        } else {
            boxes[slot * 4 + 0] = 0.f; boxes[slot * 4 + 1] = 0.f;
            boxes[slot * 4 + 2] = 0.f; boxes[slot * 4 + 3] = 0.f;
            scores[slot] = -1.f;       // matches reference's masked filler
            labels[slot] = 0;
        }
    }
}

// ---------------- pass 6: class-aware NMS + output ----------------
__global__ __launch_bounds__(1024)
void nms_kernel(const float* __restrict__ boxes, const float* __restrict__ scores,
                const int* __restrict__ labels, const int* __restrict__ pH,
                const int* __restrict__ pW, float* __restrict__ out) {
    __shared__ float work[K_TOTAL];
    __shared__ float rv[1024];
    __shared__ int   ri[1024];
    __shared__ float sOff;
    __shared__ int   keepJ[MAX_DET];
    __shared__ float keepS[MAX_DET];
    const int tid = threadIdx.x;

    // offset = labels * (max(boxes)+1)
    float m = -3.0e38f;
    for (int i = tid; i < K_TOTAL * 4; i += 1024) m = fmaxf(m, boxes[i]);
    rv[tid] = m; __syncthreads();
    for (int s = 512; s > 0; s >>= 1) {
        if (tid < s) rv[tid] = fmaxf(rv[tid], rv[tid + s]);
        __syncthreads();
    }
    if (tid == 0) sOff = rv[0] + 1.0f;
    __syncthreads();
    const float off = sOff;

    // per-thread cached offset boxes + areas (4 candidates/thread)
    float cb[4][4], car[4];
#pragma unroll
    for (int k = 0; k < 4; ++k) {
        int i = tid + k * 1024;
        if (i < K_TOTAL) {
            float o = (float)labels[i] * off;
            cb[k][0] = boxes[i * 4 + 0] + o; cb[k][1] = boxes[i * 4 + 1] + o;
            cb[k][2] = boxes[i * 4 + 2] + o; cb[k][3] = boxes[i * 4 + 3] + o;
            car[k] = (cb[k][2] - cb[k][0]) * (cb[k][3] - cb[k][1]);
            work[i] = (scores[i] > SCORE_THR) ? scores[i] : -INFINITY;
        }
    }
    __syncthreads();

    for (int t = 0; t < MAX_DET; ++t) {
        // argmax with smallest-index tiebreak (matches jnp.argmax)
        float bv = -INFINITY; int bi = 0x7fffffff;
#pragma unroll
        for (int k = 0; k < 4; ++k) {
            int i = tid + k * 1024;
            if (i < K_TOTAL) {
                float w = work[i];
                if (w > bv || (w == bv && i < bi)) { bv = w; bi = i; }
            }
        }
        rv[tid] = bv; ri[tid] = bi; __syncthreads();
        for (int s = 512; s > 0; s >>= 1) {
            if (tid < s) {
                if (rv[tid + s] > rv[tid] ||
                    (rv[tid + s] == rv[tid] && ri[tid + s] < ri[tid])) {
                    rv[tid] = rv[tid + s]; ri[tid] = ri[tid + s];
                }
            }
            __syncthreads();
        }
        const int j = ri[0]; const float sc = rv[0];
        if (tid == 0) { keepJ[t] = j; keepS[t] = sc; work[j] = -INFINITY; }
        __syncthreads();
        if (sc > -3.0e38f) {
            const float jo = (float)labels[j] * off;
            const float jx0 = boxes[j * 4 + 0] + jo, jy0 = boxes[j * 4 + 1] + jo;
            const float jx1 = boxes[j * 4 + 2] + jo, jy1 = boxes[j * 4 + 3] + jo;
            const float ja = (jx1 - jx0) * (jy1 - jy0);
#pragma unroll
            for (int k = 0; k < 4; ++k) {
                int i = tid + k * 1024;
                if (i < K_TOTAL && work[i] != -INFINITY) {
                    float lx = fmaxf(jx0, cb[k][0]), ly = fmaxf(jy0, cb[k][1]);
                    float rx = fminf(jx1, cb[k][2]), ry = fminf(jy1, cb[k][3]);
                    float iw = fmaxf(rx - lx, 0.f), ih = fmaxf(ry - ly, 0.f);
                    float inter = iw * ih;
                    float un = fmaxf(ja + car[k] - inter, 1e-9f);
                    if (inter / un > IOU_THR) work[i] = -INFINITY;
                }
            }
        }
        __syncthreads();
    }

    if (tid < MAX_DET) {
        const float sc = keepS[tid];
        const bool valid = sc > -3.0e38f;
        const int j = keepJ[tid];
        const float sx = (float)(*pW) / 1280.f;   // in_w = 160*8
        const float sy = (float)(*pH) / 1280.f;   // in_h = 160*8
        out[tid * 4 + 0] = valid ? boxes[j * 4 + 0] * sx : 0.f;
        out[tid * 4 + 1] = valid ? boxes[j * 4 + 1] * sy : 0.f;
        out[tid * 4 + 2] = valid ? boxes[j * 4 + 2] * sx : 0.f;
        out[tid * 4 + 3] = valid ? boxes[j * 4 + 3] * sy : 0.f;
        out[4 * MAX_DET + tid] = valid ? sc : 0.f;
        out[5 * MAX_DET + tid] = valid ? (float)labels[j] : -1.f;
    }
}

extern "C" void kernel_launch(void* const* d_in, const int* in_sizes, int n_in,
                              void* d_out, int out_size, void* d_ws, size_t ws_size,
                              hipStream_t stream) {
    (void)in_sizes; (void)n_in; (void)out_size; (void)ws_size;
    const float* cls[4] = {(const float*)d_in[0], (const float*)d_in[1],
                           (const float*)d_in[2], (const float*)d_in[3]};
    const float* bbx[4] = {(const float*)d_in[4], (const float*)d_in[5],
                           (const float*)d_in[6], (const float*)d_in[7]};
    const int* pH = (const int*)d_in[8];
    const int* pW = (const int*)d_in[9];

    char* ws = (char*)d_ws;
    size_t off = 0;
    auto alloc = [&](size_t bytes) {
        size_t o = off;
        off += (bytes + 255) & ~(size_t)255;
        return o;
    };
    unsigned* hist = (unsigned*)(ws + alloc(4 * 2048 * sizeof(unsigned)));
    int* meta      = (int*)(ws + alloc(256));
    int* selIdx    = (int*)(ws + alloc(K_TOTAL * sizeof(int)));
    float* selVal  = (float*)(ws + alloc(K_TOTAL * sizeof(float)));
    unsigned long long* ovKey =
        (unsigned long long*)(ws + alloc(4 * OVERCAP * sizeof(unsigned long long)));
    float* boxes   = (float*)(ws + alloc(K_TOTAL * 4 * sizeof(float)));
    float* scores  = (float*)(ws + alloc(K_TOTAL * sizeof(float)));
    int* labels    = (int*)(ws + alloc(K_TOTAL * sizeof(int)));

    static const int HWs[4] = {25600, 6400, 1600, 400};

    init_kernel<<<32, 256, 0, stream>>>(hist, meta);
    for (int l = 0; l < 4; ++l) {
        int n4 = HWs[l] * NUM_CLASSES / 4;
        int blocks = (n4 + 255) / 256; if (blocks > 1024) blocks = 1024;
        hist_kernel<<<blocks, 256, 0, stream>>>(cls[l], n4, hist + l * 2048);
    }
    scan_kernel<<<1, 32, 0, stream>>>(hist, meta);
    for (int l = 0; l < 4; ++l) {
        int n4 = HWs[l] * NUM_CLASSES / 4;
        int blocks = (n4 + 255) / 256; if (blocks > 1024) blocks = 1024;
        compact_kernel<<<blocks, 256, 0, stream>>>(cls[l], n4, HWs[l], l, meta,
                                                   selIdx, selVal, ovKey);
    }
    fixup_kernel<<<4, 1024, 0, stream>>>(meta, ovKey, selIdx, selVal);
    decode_kernel<<<32, 256, 0, stream>>>(selIdx, selVal, meta,
                                          bbx[0], bbx[1], bbx[2], bbx[3],
                                          boxes, scores, labels);
    nms_kernel<<<1, 1024, 0, stream>>>(boxes, scores, labels, pH, pW, (float*)d_out);
}
